// LilGamba_20572893348368
// MI455X (gfx1250) — matrix operand order repaired and checked by reading, not compiled, not run on hardware
//
#include <hip/hip_runtime.h>
#include <hip/hip_bf16.h>

// ---------------------------------------------------------------------------
// LilGamba forward for MI455X (gfx1250).
//  - All GEMMs: v_wmma_f32_16x16x32_bf16 (bf16 in, f32 accumulate).
//  - Weights AND activations pre-converted to bf16; GEMM staging is a pure
//    copy done with global_load_async_to_lds_b128 (ASYNCcnt) on interior tiles.
//  - Selective scan parallelized over (d,n) with 16-lane shuffle reduction.
// ---------------------------------------------------------------------------

#define D_MODEL 768
#define D_INNER 1536
#define DT_RANK 48
#define D_STATE 16
#define SEQ_L   2048
#define VOCAB_N 50264

typedef __attribute__((ext_vector_type(16))) __bf16 v16bf;
typedef __attribute__((ext_vector_type(8)))  __bf16 v8bf;
typedef __attribute__((ext_vector_type(8)))  float  v8f;

__device__ __forceinline__ unsigned short f2bf(float f) {
  unsigned int u = __float_as_uint(f);
  u += 0x7FFFu + ((u >> 16) & 1u);          // round to nearest even
  return (unsigned short)(u >> 16);
}

// async 16B copy global -> LDS (CDNA5).  lds_addr = allocation-relative LDS
// byte offset (low 32 bits of the flat shared-pointer address).
__device__ __forceinline__ void async_copy16(unsigned lds_addr, const void* gaddr) {
  asm volatile("global_load_async_to_lds_b128 %0, %1, off"
               :: "v"(lds_addr), "v"(gaddr) : "memory");
}
__device__ __forceinline__ void wait_async() {
  asm volatile("s_wait_asynccnt 0x0" ::: "memory");
}
__device__ __forceinline__ unsigned lds_off(const void* p) {
  return (unsigned)(unsigned long long)p;
}

// ------------------------------ embedding ----------------------------------
__global__ void embed_kernel(const int* __restrict__ ids,
                             const float* __restrict__ emb,
                             float* __restrict__ out) {
  int t = blockIdx.x;
  long id = ids[t];
  for (int c = threadIdx.x; c < D_MODEL; c += blockDim.x)
    out[(long)t * D_MODEL + c] = emb[id * D_MODEL + c];
}

// --------------------- rmsnorm -> bf16 (GEMM A operand) --------------------
__global__ void rmsnorm_kernel(const float* __restrict__ x,
                               const float* __restrict__ w,
                               unsigned short* __restrict__ out) {
  int row = blockIdx.x;
  const float* xr = x + (long)row * D_MODEL;
  float s = 0.f;
  for (int c = threadIdx.x; c < D_MODEL; c += 256) { float v = xr[c]; s += v * v; }
  __shared__ float red[256];
  red[threadIdx.x] = s;
  __syncthreads();
  for (int o = 128; o > 0; o >>= 1) {
    if (threadIdx.x < o) red[threadIdx.x] += red[threadIdx.x + o];
    __syncthreads();
  }
  float scale = rsqrtf(red[0] / (float)D_MODEL + 1e-5f);
  for (int c = threadIdx.x; c < D_MODEL; c += 256)
    out[(long)row * D_MODEL + c] = f2bf(xr[c] * scale * w[c]);
}

// --------------- weight transpose + f32 -> bf16: [K][N] -> [N][K] ----------
__global__ void wtr_kernel(const float* __restrict__ src, unsigned short* __restrict__ dst,
                           int K, int N) {
  __shared__ unsigned short tile[32][33];
  long mo = (long)blockIdx.z * K * N;
  int k0 = blockIdx.y * 32, n0 = blockIdx.x * 32;
  int tx = threadIdx.x & 31, ty = threadIdx.x >> 5;
  for (int r = ty; r < 32; r += 8) {
    int k = k0 + r, n = n0 + tx;
    int kc = k < K ? k : K - 1, nc = n < N ? n : N - 1;
    float v = src[mo + (long)kc * N + nc];
    tile[r][tx] = f2bf((k < K && n < N) ? v : 0.f);
  }
  __syncthreads();
  for (int r = ty; r < 32; r += 8) {
    int n = n0 + r, k = k0 + tx;
    if (n < N && k < K) dst[mo + (long)n * K + k] = tile[tx][r];
  }
}

// --------------------- depthwise causal conv (k=4) + SiLU ------------------
// writes f32 (for the scan) and bf16 (GEMM A operand of xproj)
__global__ void conv_silu_kernel(const float* __restrict__ xr,   // [L][2*D_INNER]
                                 const float* __restrict__ cw,   // [D_INNER][4]
                                 const float* __restrict__ cb,   // [D_INNER]
                                 float* __restrict__ xi,
                                 unsigned short* __restrict__ xib) {
  int idx = blockIdx.x * 256 + threadIdx.x;
  if (idx >= SEQ_L * D_INNER) return;
  int t = idx / D_INNER, c = idx % D_INNER;
  float acc = cb[c];
#pragma unroll
  for (int k = 0; k < 4; ++k) {
    int tt = t - 3 + k;
    float v = xr[(long)(tt < 0 ? 0 : tt) * (2 * D_INNER) + c];
    acc += (tt >= 0) ? cw[c * 4 + k] * v : 0.f;
  }
  float y = acc / (1.0f + __expf(-acc));   // silu
  xi[idx] = y;
  xib[idx] = f2bf(y);
}

// ------------------------- f32 -> bf16 copy --------------------------------
__global__ void tobf_kernel(const float* __restrict__ src,
                            unsigned short* __restrict__ dst, int n) {
  int i = blockIdx.x * 256 + threadIdx.x;
  if (i < n) dst[i] = f2bf(src[i]);
}

// ------------------------- softplus(z + bias) ------------------------------
__global__ void softplus_bias_kernel(float* __restrict__ delta,
                                     const float* __restrict__ b) {
  int idx = blockIdx.x * 256 + threadIdx.x;
  if (idx >= SEQ_L * D_INNER) return;
  int d = idx % D_INNER;
  float v = delta[idx] + b[d];
  delta[idx] = (v > 20.0f) ? v : logf(1.0f + __expf(v));
}

// ------------------------- selective scan ----------------------------------
__global__ void scan_kernel(const float* __restrict__ u,      // [L][D_INNER]
                            const float* __restrict__ dl,     // [L][D_INNER]
                            const float* __restrict__ xdbl,   // [L][80]
                            const float* __restrict__ Alog,   // [D_INNER][16]
                            const float* __restrict__ Dp,     // [D_INNER]
                            float* __restrict__ y) {          // [L][D_INNER]
  int d = blockIdx.x * 16 + (threadIdx.x >> 4);
  int n = threadIdx.x & 15;
  float A  = -__expf(Alog[(long)d * D_STATE + n]);
  float Dd = Dp[d];
  float h = 0.f;
  for (int t = 0; t < SEQ_L; ++t) {
    float dt = dl[(long)t * D_INNER + d];
    float ut = u [(long)t * D_INNER + d];
    float Bn = xdbl[(long)t * 80 + DT_RANK + n];
    float Cn = xdbl[(long)t * 80 + DT_RANK + D_STATE + n];
    h = h * __expf(dt * A) + (dt * ut) * Bn;
    float c = h * Cn;
    c += __shfl_xor(c, 8, 16);
    c += __shfl_xor(c, 4, 16);
    c += __shfl_xor(c, 2, 16);
    c += __shfl_xor(c, 1, 16);
    if (n == 0) y[(long)t * D_INNER + d] = c + ut * Dd;
  }
}

// ---------------- gate: g = y * silu(res), bf16 out ------------------------
__global__ void gate_kernel(const float* __restrict__ y,
                            const float* __restrict__ xr,   // [L][2*D_INNER]
                            unsigned short* __restrict__ gb) {
  int idx = blockIdx.x * 256 + threadIdx.x;
  if (idx >= SEQ_L * D_INNER) return;
  int t = idx / D_INNER, d = idx % D_INNER;
  float r = xr[(long)t * (2 * D_INNER) + D_INNER + d];
  gb[idx] = f2bf(y[idx] * (r / (1.0f + __expf(-r))));
}

// -------------------------- bf16 WMMA GEMM ---------------------------------
// C[M,N] (+)= A[M,K] @ B^T ; A bf16 [M][lda]; B [N][K] bf16 (bf16B=1) or f32.
#define BM 128
#define BN 64
#define BK 32
#define LDA_S 40
#define LDB_S 40

__global__ __launch_bounds__(256)
void gemm_bf16_wmma(const unsigned short* __restrict__ A, int lda,
                    const void* __restrict__ Bv, int bf16B,
                    float* __restrict__ C, int ldc,
                    int M, int N, int K, int beta) {
  __shared__ __align__(16) unsigned short As[BM * LDA_S];
  __shared__ __align__(16) unsigned short Bs[BN * LDB_S];

  int tid  = threadIdx.x;
  int m0   = blockIdx.y * BM;
  int n0   = blockIdx.x * BN;
  int lane = tid & 31;
  int wave = tid >> 5;
  int wm = wave >> 1;          // 0..3 -> 32-row strips
  int wn = wave & 1;           // 0..1 -> 32-col strips
  int lr = lane & 15;
  int lh = lane >> 4;

  const bool mfull = (m0 + BM <= M);
  const bool nfull = (n0 + BN <= N);

  // staging thread mappings (one 16B chunk = 8 bf16)
  int arow = tid >> 2;               // 0..63 (A rows; second pass +64)
  int aq   = (tid & 3) * 8;          // A col chunk
  int brow = tid >> 2;               // 0..63 (B rows)
  int bq   = (tid & 3) * 8;          // B col chunk
  int fr   = tid >> 3;               // f32-B rows (+32 steps)
  int fc4  = (tid & 7) * 4;

  v8f acc00 = {}; v8f acc01 = {}; v8f acc10 = {}; v8f acc11 = {};

  for (int kt = 0; kt < K; kt += BK) {
    const bool kfull = (kt + BK <= K);

    // ---- stage A tile [BM][BK] bf16 ----
    if (mfull && kfull) {
      async_copy16(lds_off(&As[arow * LDA_S + aq]),
                   &A[(long)(m0 + arow) * lda + kt + aq]);
      async_copy16(lds_off(&As[(arow + 64) * LDA_S + aq]),
                   &A[(long)(m0 + arow + 64) * lda + kt + aq]);
      if (kt + BK < K)
        __builtin_prefetch(&A[(long)(m0 + arow) * lda + kt + BK + aq], 0, 1);
    } else {
#pragma unroll
      for (int rr = 0; rr < 2; ++rr) {
        int r = arow + rr * 64;
        int gr = m0 + r;
        long rb = (long)(gr < M ? gr : M - 1) * lda;
#pragma unroll
        for (int j = 0; j < 8; ++j) {
          int gk = kt + aq + j;
          unsigned short v = A[rb + (gk < K ? gk : K - 1)];
          As[r * LDA_S + aq + j] = (gr < M && gk < K) ? v : (unsigned short)0;
        }
      }
    }

    // ---- stage B tile: Bs[n][k] ----
    if (bf16B) {
      const unsigned short* B = (const unsigned short*)Bv;
      int gn = n0 + brow;
      if (nfull && kfull) {
        async_copy16(lds_off(&Bs[brow * LDB_S + bq]), &B[(long)gn * K + kt + bq]);
        if (kt + BK < K)
          __builtin_prefetch(&B[(long)gn * K + kt + BK + bq], 0, 1);
      } else {
        long rb = (long)(gn < N ? gn : N - 1) * K;
#pragma unroll
        for (int j = 0; j < 8; ++j) {
          int gk = kt + bq + j;
          unsigned short v = B[rb + (gk < K ? gk : K - 1)];
          Bs[brow * LDB_S + bq + j] = (gn < N && gk < K) ? v : (unsigned short)0;
        }
      }
    } else {
      const float* B = (const float*)Bv;
      if (nfull && kfull) {
#pragma unroll
        for (int r = fr; r < BN; r += 32) {
          float4 v = *(const float4*)&B[(long)(n0 + r) * K + kt + fc4];
          uint2 p;
          p.x = (unsigned)f2bf(v.x) | ((unsigned)f2bf(v.y) << 16);
          p.y = (unsigned)f2bf(v.z) | ((unsigned)f2bf(v.w) << 16);
          *(uint2*)&Bs[r * LDB_S + fc4] = p;
        }
        if (kt + BK < K)
          __builtin_prefetch(&B[(long)(n0 + fr) * K + kt + BK + fc4], 0, 1);
      } else {
        for (int r = fr; r < BN; r += 32) {
          int gn = n0 + r;
          long rb = (long)(gn < N ? gn : N - 1) * K;
#pragma unroll
          for (int j = 0; j < 4; ++j) {
            int gk = kt + fc4 + j;
            float v = B[rb + (gk < K ? gk : K - 1)];
            Bs[r * LDB_S + fc4 + j] = f2bf((gn < N && gk < K) ? v : 0.f);
          }
        }
      }
    }
    wait_async();
    __syncthreads();

    // ---- fragments per ISA 7.12.2 layouts ----
    int arow0 = (wm * 32 + lr) * LDA_S + lh * 8;
    int arow1 = (wm * 32 + 16 + lr) * LDA_S + lh * 8;
    v8bf a0lo = *(const v8bf*)&As[arow0];
    v8bf a0hi = *(const v8bf*)&As[arow0 + 16];
    v8bf a1lo = *(const v8bf*)&As[arow1];
    v8bf a1hi = *(const v8bf*)&As[arow1 + 16];
    v16bf a0 = __builtin_shufflevector(a0lo, a0hi, 0,1,2,3,4,5,6,7,8,9,10,11,12,13,14,15);
    v16bf a1 = __builtin_shufflevector(a1lo, a1hi, 0,1,2,3,4,5,6,7,8,9,10,11,12,13,14,15);

    int brow0 = (wn * 32 + lr) * LDB_S + lh * 8;
    int brow1 = (wn * 32 + 16 + lr) * LDB_S + lh * 8;
    v8bf b0lo = *(const v8bf*)&Bs[brow0];
    v8bf b0hi = *(const v8bf*)&Bs[brow0 + 16];
    v8bf b1lo = *(const v8bf*)&Bs[brow1];
    v8bf b1hi = *(const v8bf*)&Bs[brow1 + 16];
    v16bf b0 = __builtin_shufflevector(b0lo, b0hi, 0,1,2,3,4,5,6,7,8,9,10,11,12,13,14,15);
    v16bf b1 = __builtin_shufflevector(b1lo, b1hi, 0,1,2,3,4,5,6,7,8,9,10,11,12,13,14,15);

    acc00 = __builtin_amdgcn_wmma_f32_16x16x32_bf16(false, a0, false, b0, (short)0, acc00, false, false);
    acc01 = __builtin_amdgcn_wmma_f32_16x16x32_bf16(false, a0, false, b1, (short)0, acc01, false, false);
    acc10 = __builtin_amdgcn_wmma_f32_16x16x32_bf16(false, a1, false, b0, (short)0, acc10, false, false);
    acc11 = __builtin_amdgcn_wmma_f32_16x16x32_bf16(false, a1, false, b1, (short)0, acc11, false, false);
    __syncthreads();
  }

  // ---- epilogue ----
  auto store_tile = [&](v8f acc, int mt, int nt) {
    int baseM = m0 + wm * 32 + mt * 16 + lh * 8;
    int baseN = n0 + wn * 32 + nt * 16 + lr;
    if (mfull && nfull) {
#pragma unroll
      for (int r = 0; r < 8; ++r) {
        long o = (long)(baseM + r) * ldc + baseN;
        float v = acc[r];
        if (beta) v += C[o];
        C[o] = v;
      }
    } else {
#pragma unroll
      for (int r = 0; r < 8; ++r) {
        int row = baseM + r, col = baseN;
        if (row < M && col < N) {
          float v = acc[r];
          if (beta) v += C[(long)row * ldc + col];
          C[(long)row * ldc + col] = v;
        }
      }
    }
  };
  store_tile(acc00, 0, 0);
  store_tile(acc01, 0, 1);
  store_tile(acc10, 1, 0);
  store_tile(acc11, 1, 1);
}

static inline void gemm(hipStream_t s, const unsigned short* A, int lda,
                        const void* B, int bf16B,
                        float* C, int ldc, int M, int N, int K, int beta) {
  dim3 g((N + BN - 1) / BN, (M + BM - 1) / BM);
  gemm_bf16_wmma<<<g, 256, 0, s>>>(A, lda, B, bf16B, C, ldc, M, N, K, beta);
}

// ---------------------------------------------------------------------------
extern "C" void kernel_launch(void* const* d_in, const int* in_sizes, int n_in,
                              void* d_out, int out_size, void* d_ws, size_t ws_size,
                              hipStream_t stream) {
  const int*   ids      = (const int*)d_in[0];
  const float* emb      = (const float*)d_in[1];
  const float* inproj   = (const float*)d_in[2];   // [16][768][3072]
  const float* convw    = (const float*)d_in[3];   // [16][1536][1][4]
  const float* convb    = (const float*)d_in[4];   // [16][1536]
  const float* xprojw   = (const float*)d_in[5];   // [16][1536][80]
  const float* dtprojw  = (const float*)d_in[6];   // [16][48][1536]
  const float* dtprojb  = (const float*)d_in[7];   // [16][1536]
  const float* Alog     = (const float*)d_in[8];   // [16][1536][16]
  const float* Dpar     = (const float*)d_in[9];   // [16][1536]
  const float* outprojw = (const float*)d_in[10];  // [16][1536][768]
  const float* normw    = (const float*)d_in[11];  // [4][768]
  const float* normfw   = (const float*)d_in[12];  // [768]
  float* out = (float*)d_out;

  const long S = (long)SEQ_L * D_MODEL;            // 1,572,864 floats
  // nrm_bf must be OUTSIDE d_out (read while the final GEMM writes d_out)
  unsigned short* nrm_bf = (unsigned short*)d_ws;  // 3.1 MB in ws

  // ---- scratch arena at the tail of d_out (dead before final GEMM) ----
  const long WI = 16L * 3072 * 768;   // inprojT  ushorts
  const long WX = 16L * 80 * 1536;    // xprojT
  const long WD = 16L * 1536 * 48;    // dtprojT
  const long WO = 16L * 768 * 1536;   // outprojT
  const long XD = (long)SEQ_L * 80;
  const long actF = 15 * S + XD;                          // f32 activations
  const long actB = S + S + XD / 2 + S / 2;               // bf16 acts (floats)
  const long need = actF + actB + (WI + WX + WD + WO) / 2;  // ~57.7M floats
  float* base = out + ((long)out_size - need);
  float* Xh[5];
  for (int i = 0; i < 5; ++i) Xh[i] = base + (long)i * S;
  float* xr    = base + 5 * S;                     // [L][3072]
  float* xi    = base + 9 * S;                     // [L][1536]
  float* delta = base + 11 * S;                    // [L][1536]
  float* yb    = base + 13 * S;                    // [L][1536]
  float* xdbl  = base + 15 * S;                    // [L][80]
  unsigned short* xi_bf   = (unsigned short*)(base + actF);        // [L][1536]
  unsigned short* g_bf    = xi_bf + 2 * S;                         // [L][1536]
  unsigned short* xdbl_bf = g_bf + 2 * S;                          // [L][80]
  unsigned short* inprojT  = xdbl_bf + XD;
  unsigned short* xprojT   = inprojT + WI;
  unsigned short* dtprojT  = xprojT + WX;
  unsigned short* outprojT = dtprojT + WD;

  // ---- pre-transpose + convert all inner weights to bf16 [N][K] ----
  wtr_kernel<<<dim3(96, 24, 16), 256, 0, stream>>>(inproj,   inprojT,  D_MODEL, 2 * D_INNER);
  wtr_kernel<<<dim3( 3, 48, 16), 256, 0, stream>>>(xprojw,   xprojT,   D_INNER, 80);
  wtr_kernel<<<dim3(48,  2, 16), 256, 0, stream>>>(dtprojw,  dtprojT,  DT_RANK, D_INNER);
  wtr_kernel<<<dim3(24, 48, 16), 256, 0, stream>>>(outprojw, outprojT, D_INNER, D_MODEL);

  embed_kernel<<<SEQ_L, 256, 0, stream>>>(ids, emb, Xh[0]);

  const int EW = (SEQ_L * D_INNER) / 256;          // 12288 elementwise blocks

  for (int li = 0; li < 4; ++li) {
    hipMemcpyAsync(Xh[li + 1], Xh[li], S * sizeof(float),
                   hipMemcpyDeviceToDevice, stream);   // residual = queue[0]
    int nb = (li + 1 < 4) ? (li + 1) : 4;
    for (int i = 0; i < nb; ++i) {
      long gi = (long)li * 4 + i;
      rmsnorm_kernel<<<SEQ_L, 256, 0, stream>>>(Xh[li - i], normw + (long)li * D_MODEL, nrm_bf);
      // xr = normed @ in_proj            (N=3072, K=768)
      gemm(stream, nrm_bf, D_MODEL, inprojT + gi * (long)(2 * D_INNER) * D_MODEL, 1,
           xr, 2 * D_INNER, SEQ_L, 2 * D_INNER, D_MODEL, 0);
      conv_silu_kernel<<<EW, 256, 0, stream>>>(xr, convw + gi * D_INNER * 4,
                                               convb + gi * D_INNER, xi, xi_bf);
      // x_dbl = xi @ xproj               (N=80, K=1536)
      gemm(stream, xi_bf, D_INNER, xprojT + gi * 80L * D_INNER, 1,
           xdbl, 80, SEQ_L, 80, D_INNER, 0);
      tobf_kernel<<<(int)((XD + 255) / 256), 256, 0, stream>>>(xdbl, xdbl_bf, (int)XD);
      // delta = x_dbl[:, :48] @ dtproj   (N=1536, K=48)
      gemm(stream, xdbl_bf, 80, dtprojT + gi * (long)D_INNER * DT_RANK, 1,
           delta, D_INNER, SEQ_L, D_INNER, DT_RANK, 0);
      softplus_bias_kernel<<<EW, 256, 0, stream>>>(delta, dtprojb + gi * D_INNER);
      scan_kernel<<<D_INNER / 16, 256, 0, stream>>>(xi, delta, xdbl,
                                                    Alog + gi * D_INNER * D_STATE,
                                                    Dpar + gi * D_INNER, yb);
      gate_kernel<<<EW, 256, 0, stream>>>(yb, xr, g_bf);
      // x_new += g @ outproj             (N=768, K=1536)
      gemm(stream, g_bf, D_INNER, outprojT + gi * (long)D_MODEL * D_INNER, 1,
           Xh[li + 1], D_MODEL, SEQ_L, D_MODEL, D_INNER, 1);
    }
  }

  // logits = rmsnorm(x, normf) @ emb.T  (emb is [VOCAB][768] f32 == [N][K])
  rmsnorm_kernel<<<SEQ_L, 256, 0, stream>>>(Xh[4], normfw, nrm_bf);
  gemm(stream, nrm_bf, D_MODEL, emb, 0, out, VOCAB_N, SEQ_L, VOCAB_N, D_MODEL, 0);
}
